// LSTMGenerator_12257836662794
// MI455X (gfx1250) — matrix operand order repaired and checked by reading, not compiled, not run on hardware
//
#include <hip/hip_runtime.h>

#define VOCAB 18
#define EMB 200
#define HID 200
#define SEQ 128
#define BATCH 2048
#define GATES 800        // 4*HID
#define KPAD 224         // HID padded to 7*32 (K dim of WMMA)
#define KC 7             // k-chunks of 32
#define MT 16            // batch rows per workgroup (one WMMA M tile)
#define NWAVES 7
#define THREADS (NWAVES * 32)
#define NHT 14           // hidden tiles covered (224 units, valid < 200)
#define WHH_TILES (4 * NHT)   // 56 B-tiles for recurrent weights
#define WOUT_TILES 2          // vocab padded to 32

#define HN_OFF (BATCH * SEQ * VOCAB)          // 4718592
#define CN_OFF (HN_OFF + BATCH * HID)         // 5128192

typedef __attribute__((ext_vector_type(16))) __bf16 v16bf;
typedef __attribute__((ext_vector_type(8)))  float  v8f;

union AFrag { v16bf v; unsigned int u[8]; };
union BFrag { v16bf v; uint4 q[2]; };

__device__ __forceinline__ unsigned short f32_to_bf16(float f) {
    union { float f; unsigned int u; } c; c.f = f;
    unsigned int r = c.u + 0x7FFFu + ((c.u >> 16) & 1u);   // round-nearest-even
    return (unsigned short)(r >> 16);
}
__device__ __forceinline__ float fast_sig(float x) {
    return __builtin_amdgcn_rcpf(1.0f + __expf(-x));
}
__device__ __forceinline__ float fast_tanh(float x) {
    float e = __expf(-2.0f * x);
    return (1.0f - e) * __builtin_amdgcn_rcpf(1.0f + e);
}
__device__ __forceinline__ v8f vzero8() {
    v8f z = {0.f, 0.f, 0.f, 0.f, 0.f, 0.f, 0.f, 0.f};
    return z;
}

// ---------------------------------------------------------------------------
// Prologue 1: gate table  G[v][g] = E_masked[v] . W_ih[g] + b_ih[g] + b_hh[g]
// (collapses the 84-GFLOP input projection into an 18x800 lookup table)
// ---------------------------------------------------------------------------
__global__ void prep_gtab(const float* __restrict__ E, const float* __restrict__ Wih,
                          const float* __restrict__ bih, const float* __restrict__ bhh,
                          float* __restrict__ Gtab)
{
    int tid = blockIdx.x * blockDim.x + threadIdx.x;
    if (tid >= VOCAB * GATES) return;
    int v = tid / GATES, g = tid % GATES;
    float acc = bih[g] + bhh[g];
    if (v != 0) {                      // padding_idx = 0 -> zero embedding row
        for (int k = 0; k < EMB; ++k)
            acc += E[v * EMB + k] * Wih[g * EMB + k];
    }
    Gtab[tid] = acc;
}

// ---------------------------------------------------------------------------
// Prologue 2: pack W_hh (f32, row-major [800][200]) into bf16 WMMA B-fragment
// order: dst[((tile*7+kc)*32+lane)*16+e], tile = gate*14 + hidden_tile.
// B k order used here (contiguous, same as A): lanes 0-15 N=lane K=0..15, lanes 16-31 N=lane-16
// K=16..31, 2 bf16 per VGPR sequential -> element e maps to k=(lane>>4)*16+e.
// ---------------------------------------------------------------------------
__global__ void pack_whh(const float* __restrict__ Whh, unsigned short* __restrict__ dst)
{
    int tid = blockIdx.x * blockDim.x + threadIdx.x;
    if (tid >= WHH_TILES * KC * 512) return;
    int e    = tid & 15;
    int lane = (tid >> 4) & 31;
    int kc   = (tid >> 9) % KC;
    int tile = tid / (KC * 512);
    int k  = kc * 32 + (lane >> 4) * 16 + e;
    int gt = tile / NHT, ht = tile % NHT;
    int hu = ht * 16 + (lane & 15);
    float v = (k < HID && hu < HID) ? Whh[(gt * HID + hu) * HID + k] : 0.0f;
    dst[tid] = f32_to_bf16(v);
}

// Prologue 3: pack W_out (f32 [18][200]) the same way, 2 N-tiles (vocab -> 32)
__global__ void pack_wout(const float* __restrict__ Wout, unsigned short* __restrict__ dst)
{
    int tid = blockIdx.x * blockDim.x + threadIdx.x;
    if (tid >= WOUT_TILES * KC * 512) return;
    int e    = tid & 15;
    int lane = (tid >> 4) & 31;
    int kc   = (tid >> 9) % KC;
    int tile = tid / (KC * 512);
    int k  = kc * 32 + (lane >> 4) * 16 + e;
    int vv = tile * 16 + (lane & 15);
    float v = (k < HID && vv < VOCAB) ? Wout[vv * HID + k] : 0.0f;
    dst[tid] = f32_to_bf16(v);
}

// ---------------------------------------------------------------------------
// Main recurrent kernel. 128 workgroups x 224 threads (7 waves, wave32).
// Each WG owns 16 batch rows; h double-buffered bf16 in LDS, c in registers.
// Per timestep: 56 v_wmma_f32_16x16x32_bf16 per wave (+7 for inline logits).
// Packed weights are re-streamed from L2 each timestep (they are laundered
// through an empty asm so LICM cannot hoist 63x16 VGPRs of fragments and
// spill them to scratch, which is what the previous build did).
// ---------------------------------------------------------------------------
__global__ __launch_bounds__(THREADS) void lstm_main(
    const int* __restrict__ x, const float* __restrict__ Gtab,
    const unsigned short* __restrict__ Bhh, const unsigned short* __restrict__ Bout,
    const float* __restrict__ b_out, float* __restrict__ out)
{
    __shared__ int            tokLds[MT * SEQ];
    __shared__ unsigned short hbuf[2 * MT * KPAD];

    const int tid     = threadIdx.x;
    const int lane    = tid & 31;
    const int wave    = tid >> 5;
    const int nlane   = lane & 15;
    const int laneHi  = lane >> 4;          // 0 or 1 (half-wave)
    const int mbase   = laneHi * 8;
    const int rowbase = blockIdx.x * MT;

    // Stage this tile's tokens and zero both h buffers (padded cols stay 0).
    for (int i = tid; i < MT * SEQ; i += THREADS) {
        int r = i >> 7, tt = i & (SEQ - 1);
        tokLds[i] = x[(rowbase + r) * SEQ + tt];
    }
    for (int i = tid; i < 2 * MT * KPAD; i += THREADS) hbuf[i] = 0;
    __syncthreads();

    float bias = 0.0f;
    if (wave < 2) {
        int v = wave * 16 + nlane;
        if (v < VOCAB) bias = b_out[v];
    }

    const int  ht0    = wave * 2;               // this wave's hidden tiles
    const int  hu0    = ht0 * 16 + nlane;
    const int  hu1    = hu0 + 16;
    const bool valid0 = hu0 < HID;
    const bool valid1 = hu1 < HID;

    v8f cst[2];
    cst[0] = vzero8();
    cst[1] = vzero8();

    #pragma unroll 1
    for (int t = 0; t < SEQ; ++t) {
        // Launder the packed-weight base pointers: forces the B-fragment
        // loads to stay inside the loop (L2-resident stream) instead of
        // being hoisted into ~1000 VGPRs and spilled to scratch.
        unsigned int lz = 0;
        asm volatile("" : "+v"(lz));
        const unsigned short* __restrict__ BhhT  = Bhh + lz;
        const unsigned short* __restrict__ BoutT = Bout + lz;

        const int p = t & 1;
        const unsigned short* hread  = hbuf + p * (MT * KPAD);
        unsigned short*       hwrite = hbuf + (1 - p) * (MT * KPAD);

        int tk[8];
        #pragma unroll
        for (int j = 0; j < 8; ++j) tk[j] = tokLds[(mbase + j) * SEQ + t];

        // Init gate accumulators from the precomputed input-projection table
        // (C/D layout: vgpr j -> M = laneHi*8 + j, N = nlane).
        v8f acc[4][2];
        #pragma unroll
        for (int gt = 0; gt < 4; ++gt) {
            #pragma unroll
            for (int s = 0; s < 2; ++s) {
                const int  hu    = s ? hu1 : hu0;
                const bool valid = s ? valid1 : valid0;
                const int  col   = gt * HID + hu;
                #pragma unroll
                for (int j = 0; j < 8; ++j)
                    acc[gt][s][j] = valid ? Gtab[tk[j] * GATES + col] : 0.0f;
            }
        }

        // gates += h_{t-1} @ W_hh^T   (bf16 WMMA, K padded to 224)
        #pragma unroll
        for (int kc = 0; kc < KC; ++kc) {
            AFrag a;
            #pragma unroll
            for (int v = 0; v < 8; ++v) {
                // A layout: vgpr v holds K pair kb..kb+1; lanes>=16 add +16.
                const int kb = kc * 32 + laneHi * 16 + v * 2;
                a.u[v] = *(const unsigned int*)(hread + nlane * KPAD + kb);
            }
            #pragma unroll
            for (int gt = 0; gt < 4; ++gt) {
                #pragma unroll
                for (int s = 0; s < 2; ++s) {
                    const int tile = gt * NHT + ht0 + s;
                    const uint4* bp = (const uint4*)BhhT + ((tile * KC + kc) * 32 + lane) * 2;
                    BFrag b;
                    b.q[0] = bp[0];
                    b.q[1] = bp[1];
                    acc[gt][s] = __builtin_amdgcn_wmma_f32_16x16x32_bf16(
                        false, a.v, false, b.v, (short)0, acc[gt][s], false, false);
                }
            }
        }

        // Fused LSTM cell update; h -> bf16 into the other LDS buffer.
        #pragma unroll
        for (int s = 0; s < 2; ++s) {
            const int  hu    = s ? hu1 : hu0;
            const bool valid = s ? valid1 : valid0;
            #pragma unroll
            for (int j = 0; j < 8; ++j) {
                const float ig = fast_sig(acc[0][s][j]);
                const float fg = fast_sig(acc[1][s][j]);
                const float gg = fast_tanh(acc[2][s][j]);
                const float og = fast_sig(acc[3][s][j]);
                const float cc = fg * cst[s][j] + ig * gg;
                cst[s][j] = cc;
                const float hh = og * fast_tanh(cc);
                const int m = mbase + j;
                if (valid) {
                    hwrite[m * KPAD + hu] = f32_to_bf16(hh);
                    if (t == SEQ - 1) {
                        out[HN_OFF + (size_t)(rowbase + m) * HID + hu] = hh;
                        out[CN_OFF + (size_t)(rowbase + m) * HID + hu] = cc;
                    }
                }
            }
        }
        __syncthreads();

        // Inline output head: logits[b,t,:] = h_t @ W_out^T + b_out.
        // Waves 0/1 each own 16 vocab columns (vocab padded 18 -> 32).
        if (wave < 2) {
            v8f lacc = vzero8();
            #pragma unroll
            for (int kc = 0; kc < KC; ++kc) {
                AFrag a;
                #pragma unroll
                for (int v = 0; v < 8; ++v) {
                    const int kb = kc * 32 + laneHi * 16 + v * 2;
                    a.u[v] = *(const unsigned int*)(hwrite + nlane * KPAD + kb);
                }
                const uint4* bp = (const uint4*)BoutT + ((wave * KC + kc) * 32 + lane) * 2;
                BFrag b;
                b.q[0] = bp[0];
                b.q[1] = bp[1];
                lacc = __builtin_amdgcn_wmma_f32_16x16x32_bf16(
                    false, a.v, false, b.v, (short)0, lacc, false, false);
            }
            const int v = wave * 16 + nlane;
            if (v < VOCAB) {
                #pragma unroll
                for (int j = 0; j < 8; ++j) {
                    const int m = mbase + j;
                    out[((size_t)(rowbase + m) * SEQ + t) * VOCAB + v] = lacc[j] + bias;
                }
            }
        }
    }
}

// ---------------------------------------------------------------------------
extern "C" void kernel_launch(void* const* d_in, const int* in_sizes, int n_in,
                              void* d_out, int out_size, void* d_ws, size_t ws_size,
                              hipStream_t stream) {
    (void)in_sizes; (void)n_in; (void)out_size; (void)ws_size;

    const int*   x    = (const int*)  d_in[0];
    const float* E    = (const float*)d_in[1];
    const float* Wih  = (const float*)d_in[2];
    const float* Whh  = (const float*)d_in[3];
    const float* bih  = (const float*)d_in[4];
    const float* bhh  = (const float*)d_in[5];
    const float* Wout = (const float*)d_in[6];
    const float* bout = (const float*)d_in[7];
    float* out = (float*)d_out;

    // Workspace layout: packed bf16 W_hh | packed bf16 W_out | f32 gate table
    unsigned short* Bhh  = (unsigned short*)d_ws;                 // 56*7*512 u16
    unsigned short* Bout = Bhh + WHH_TILES * KC * 512;            //  2*7*512 u16
    float*          Gtab = (float*)(Bout + WOUT_TILES * KC * 512);// 18*800 f32

    {
        int n = WHH_TILES * KC * 512;
        hipLaunchKernelGGL(pack_whh, dim3((n + 255) / 256), dim3(256), 0, stream, Whh, Bhh);
    }
    {
        int n = WOUT_TILES * KC * 512;
        hipLaunchKernelGGL(pack_wout, dim3((n + 255) / 256), dim3(256), 0, stream, Wout, Bout);
    }
    {
        int n = VOCAB * GATES;
        hipLaunchKernelGGL(prep_gtab, dim3((n + 255) / 256), dim3(256), 0, stream,
                           E, Wih, bih, bhh, Gtab);
    }
    hipLaunchKernelGGL(lstm_main, dim3(BATCH / MT), dim3(THREADS), 0, stream,
                       x, Gtab, Bhh, Bout, bout, out);
}
